// EnhancedPointNetPlusPlusClassifier_60550448939443
// MI455X (gfx1250) — matrix operand order repaired and checked by reading, not compiled, not run on hardware
//
#include <hip/hip_runtime.h>

// ---------------------------------------------------------------------------
// PointNet++ classifier for gfx1250 (MI455X)
// bf16 WMMA GEMMs (2x2 register-blocked) + fused BN/ReLU (+ fused maxpool)
// ---------------------------------------------------------------------------

typedef __attribute__((ext_vector_type(16))) __bf16          v16bf;
typedef __attribute__((ext_vector_type(8)))  float           v8f;
typedef __attribute__((ext_vector_type(16))) unsigned short  v16u;
typedef __attribute__((ext_vector_type(8)))  unsigned short  v8u;

#define kB 32        // batch
#define kN 4096      // points

__device__ __forceinline__ unsigned short f32_to_bf16(float f) {
    unsigned int u = __float_as_uint(f);
    u += 0x7FFFu + ((u >> 16) & 1u);            // round-to-nearest-even
    return (unsigned short)(u >> 16);
}
__device__ __forceinline__ float bf16_to_f32(unsigned short u) {
    return __uint_as_float(((unsigned int)u) << 16);
}

// ---------------------------------------------------------------------------
// (B,3,N) -> (B,N,3)
__global__ void transpose_xyz_kernel(const float* __restrict__ in,
                                     float* __restrict__ out, int total) {
    int i = blockIdx.x * blockDim.x + threadIdx.x;
    if (i >= total) return;
    int b = i / (kN * 3);
    int r = i - b * kN * 3;
    int n = r / 3, c = r - n * 3;
    out[i] = in[(size_t)b * 3 * kN + (size_t)c * kN + n];
}

// ---------------------------------------------------------------------------
// Farthest point sampling: one block per batch, iterative argmax via LDS tree.
__global__ void fps_kernel(const float* __restrict__ xyz, int* __restrict__ outIdx,
                           int N, int npoint) {
    const int b = blockIdx.x, tid = threadIdx.x;
    const float* p = xyz + (size_t)b * N * 3;
    __shared__ float s_best[256];
    __shared__ int   s_bidx[256];
    __shared__ int   s_far;
    float dist[16];
    const int per = (N + 255) / 256;
    for (int t = 0; t < per; ++t) dist[t] = 1e10f;
    if (tid == 0) s_far = 0;
    __syncthreads();
    for (int it = 0; it < npoint; ++it) {
        int far = s_far;
        if (tid == 0) outIdx[b * npoint + it] = far;
        float cx = p[far * 3 + 0], cy = p[far * 3 + 1], cz = p[far * 3 + 2];
        float best = -1.0f; int bi = 0;
        for (int t = 0; t < per; ++t) {
            int n = t * 256 + tid;
            if (n < N) {
                float dx = p[n * 3 + 0] - cx, dy = p[n * 3 + 1] - cy, dz = p[n * 3 + 2] - cz;
                float d = dx * dx + dy * dy + dz * dz;
                float dn = fminf(dist[t], d);
                dist[t] = dn;
                if (dn > best) { best = dn; bi = n; }
            }
        }
        s_best[tid] = best; s_bidx[tid] = bi;
        __syncthreads();
        for (int off = 128; off > 0; off >>= 1) {
            if (tid < off && s_best[tid + off] > s_best[tid]) {
                s_best[tid] = s_best[tid + off]; s_bidx[tid] = s_bidx[tid + off];
            }
            __syncthreads();
        }
        if (tid == 0) s_far = s_bidx[0];
        __syncthreads();
    }
}

// gather sampled centers: out[b,s,:] = xyz[b, idx[b,s], :]
__global__ void gather_xyz_kernel(const float* __restrict__ xyz, const int* __restrict__ idx,
                                  float* __restrict__ out, int N, int total) {
    int i = blockIdx.x * blockDim.x + threadIdx.x;
    if (i >= total) return;
    int b = i / (total / kB);           // total = kB * S
    int id = idx[i];
    const float* p = xyz + ((size_t)b * N + id) * 3;
    float* o = out + (size_t)i * 3;
    o[0] = p[0]; o[1] = p[1]; o[2] = p[2];
}

// ---------------------------------------------------------------------------
// Ball query + grouping, fused into bf16 GEMM-A layout (wave32 ballot scan).
__global__ void ball_group_kernel(const float* __restrict__ sx,      // (B,Nsrc,3)
                                  const float* __restrict__ ctr,     // (B,S,3)
                                  const float* __restrict__ featF,   // f32 feats or null
                                  const unsigned short* __restrict__ featB, // bf16 feats or null
                                  int Cf, unsigned short* __restrict__ G,
                                  int Nsrc, int S, int ns, int Kpad, float r2) {
    const int g = blockIdx.x;                   // over B*S
    const int b = g / S;
    const int tid = threadIdx.x;                // blockDim == ns (32 or 64)
    const float* base = sx + (size_t)b * Nsrc * 3;
    const float cx = ctr[g * 3 + 0], cy = ctr[g * 3 + 1], cz = ctr[g * 3 + 2];

    __shared__ int s_sel[64];
    __shared__ int s_cnt;
    if (tid == 0) { s_sel[0] = 0; }
    __syncthreads();

    if (tid < 32) {                             // one wave does the ordered scan
        int cnt = 0;
        for (int j0 = 0; j0 < Nsrc && cnt < ns; j0 += 32) {
            int j = j0 + tid;
            bool in = false;
            if (j < Nsrc) {
                float dx = base[j * 3 + 0] - cx;
                float dy = base[j * 3 + 1] - cy;
                float dz = base[j * 3 + 2] - cz;
                in = (dx * dx + dy * dy + dz * dz) <= r2;
            }
            unsigned long long mask = __ballot(in);
            if (in) {
                int pos = cnt + __popcll(mask & ((1ull << tid) - 1ull));
                if (pos < ns) s_sel[pos] = j;
            }
            cnt += __popcll(mask);
        }
        if (tid == 0) s_cnt = cnt < ns ? cnt : ns;
    }
    __syncthreads();

    const int cnt = s_cnt;
    const int k = tid;
    const int id = (k < cnt && cnt > 0) ? s_sel[k] : s_sel[0];
    unsigned short* go = G + ((size_t)g * ns + k) * Kpad;
    go[0] = f32_to_bf16(base[id * 3 + 0] - cx);
    go[1] = f32_to_bf16(base[id * 3 + 1] - cy);
    go[2] = f32_to_bf16(base[id * 3 + 2] - cz);
    if (featF) {
        for (int c = 0; c < Cf; ++c)
            go[3 + c] = f32_to_bf16(featF[((size_t)b * Nsrc + id) * Cf + c]);
    } else {
        const unsigned short* f = featB + ((size_t)b * Nsrc + id) * Cf;
        for (int c = 0; c < Cf; ++c) go[3 + c] = f[c];
    }
    for (int c = 3 + Cf; c < Kpad; ++c) go[c] = 0;
}

// group_all for SA3
__global__ void group_all_kernel(const float* __restrict__ nx,
                                 const unsigned short* __restrict__ feat,
                                 unsigned short* __restrict__ G,
                                 int Sp, int Cf, int Kpad) {
    const int b = blockIdx.x, j = threadIdx.x;   // block = Sp
    const size_t row = (size_t)b * Sp + j;
    unsigned short* go = G + row * Kpad;
    const float* x = nx + row * 3;
    go[0] = f32_to_bf16(x[0]); go[1] = f32_to_bf16(x[1]); go[2] = f32_to_bf16(x[2]);
    const unsigned short* f = feat + row * Cf;
    for (int c = 0; c < Cf; ++c) go[3 + c] = f[c];
    for (int c = 3 + Cf; c < Kpad; ++c) go[c] = 0;
}

// ---------------------------------------------------------------------------
// Per-layer prep: bf16-pack weights (zero-padded to Kpad) and fold
// conv-bias + BN into (scale, shift):  out = relu(dot*scale + shift)
__global__ void prep_layer_kernel(const float* __restrict__ W, const float* __restrict__ bias,
                                  const float* __restrict__ gm, const float* __restrict__ be,
                                  const float* __restrict__ mean, const float* __restrict__ var,
                                  unsigned short* __restrict__ Wb,
                                  float* __restrict__ scale, float* __restrict__ shift,
                                  int Cin, int Kpad) {
    const int o = blockIdx.x;                    // grid = Cout
    if (threadIdx.x == 0) {
        float s = gm[o] * rsqrtf(var[o] + 1e-5f);
        scale[o] = s;
        shift[o] = (bias[o] - mean[o]) * s + be[o];
    }
    for (int k = threadIdx.x; k < Kpad; k += blockDim.x)
        Wb[(size_t)o * Kpad + k] = (k < Cin) ? f32_to_bf16(W[(size_t)o * Cin + k]) : 0;
}

// ---------------------------------------------------------------------------
// WMMA fragment loads (CDNA5 ISA §7.12.2 wave32 layouts)
__device__ __forceinline__ v16bf load_frag_a(const unsigned short* p) {
    // A 16x32 bf16: lane(l16,h) holds K = 16h..16h+15 (contiguous)
    v8u a0 = *(const v8u*)(p);
    v8u a1 = *(const v8u*)(p + 8);
    v16u au = __builtin_shufflevector(a0, a1, 0, 1, 2, 3, 4, 5, 6, 7,
                                              8, 9, 10, 11, 12, 13, 14, 15);
    return __builtin_bit_cast(v16bf, au);
}
__device__ __forceinline__ v16bf load_frag_b(const unsigned short* p) {
    // B 32x16 bf16: B[k][n] = Wt[n][k]; lane holds K = 16h..16h+15 (contiguous)
    return __builtin_bit_cast(v16bf, *(const v16u*)(p));
}
__device__ __forceinline__ v8f wmma_bf16(v16bf a, v16bf b, v8f c) {
    return __builtin_amdgcn_wmma_f32_16x16x32_bf16(false, a, false, b,
                                                   (short)0, c, false, false);
}

// ---------------------------------------------------------------------------
// 2x2 register-blocked bf16 WMMA GEMM, fused BN+ReLU epilogue.
// Block = 256 threads = 8 waves arranged 4(M) x 2(N); each wave owns a 32x32
// D tile (4 accumulators, 4 WMMAs per 32-K step -> 2x the FLOP/byte of 1-tile).
// POOL=true additionally max-pools groups of `ns` consecutive rows entirely
// on-chip (per-lane j-max -> shfl_xor(16) across EXEC halves -> LDS across
// waves sharing a group), writing pooled (M/ns) x N output and skipping the
// big intermediate round-trip through HBM.
template <bool POOL>
__global__ void __launch_bounds__(256)
wmma_gemm_kernel(const unsigned short* __restrict__ A,   // M x K (bf16 bits)
                 const unsigned short* __restrict__ Wt,  // N x K (bf16 bits)
                 const float* __restrict__ scale,
                 const float* __restrict__ shift,
                 unsigned short* __restrict__ out,       // (M or M/ns) x N
                 int M, int K, int N, int ns) {
    const int tid  = threadIdx.x;
    const int wave = tid >> 5;
    const int lane = tid & 31;
    const int half = lane >> 4;
    const int l16  = lane & 15;
    const int mw   = wave & 3;          // M sub-block within the 128-row block
    const int nw   = wave >> 2;         // N sub-block within the 64-col block
    const int mBase = (blockIdx.x * 4 + mw) * 32;
    const int nBase = blockIdx.y * 64 + nw * 32;

    __shared__ float s_part[4][64];     // [mw][block-local col] pooled partials

    if (!POOL) {
        if (mBase >= M) return;         // wave-uniform skip (EXEC stays all-ones)
    }

    const size_t rowA0 = (size_t)(mBase + l16) * K;
    const size_t rowB0 = (size_t)(nBase + l16) * K;
    const size_t strK16 = (size_t)16 * K;

    v8f acc00 = {}, acc01 = {}, acc10 = {}, acc11 = {};
    for (int k0 = 0; k0 < K; k0 += 32) {
        const unsigned short* pA = A + rowA0 + k0 + half * 16;
        const unsigned short* pB = Wt + rowB0 + k0 + half * 16;
        __builtin_prefetch(pA + 256, 0, 0);               // global_prefetch_b8
        v16bf a0 = load_frag_a(pA);
        v16bf a1 = load_frag_a(pA + strK16);
        v16bf b0 = load_frag_b(pB);
        v16bf b1 = load_frag_b(pB + strK16);
        acc00 = wmma_bf16(a0, b0, acc00);
        acc01 = wmma_bf16(a0, b1, acc01);
        acc10 = wmma_bf16(a1, b0, acc10);
        acc11 = wmma_bf16(a1, b1, acc11);
    }

    // D layout per subtile: VGPR j -> row = j + 8*half, col = l16
    const int n0 = nBase + l16;         // N subtile 0 column
    const int n1 = nBase + 16 + l16;    // N subtile 1 column
    const float sc0 = scale[n0], sh0 = shift[n0];
    const float sc1 = scale[n1], sh1 = shift[n1];

    if (!POOL) {
        unsigned short* p0 = out + (size_t)(mBase + half * 8) * N;
        unsigned short* p1 = out + (size_t)(mBase + 16 + half * 8) * N;
#pragma unroll
        for (int j = 0; j < 8; ++j) {
            p0[(size_t)j * N + n0] = f32_to_bf16(fmaxf(acc00[j] * sc0 + sh0, 0.0f));
            p0[(size_t)j * N + n1] = f32_to_bf16(fmaxf(acc01[j] * sc1 + sh1, 0.0f));
            p1[(size_t)j * N + n0] = f32_to_bf16(fmaxf(acc10[j] * sc0 + sh0, 0.0f));
            p1[(size_t)j * N + n1] = f32_to_bf16(fmaxf(acc11[j] * sc1 + sh1, 0.0f));
        }
    } else {
        // per-lane max over the 16 rows this (lane,half) owns within the wave's
        // 32-row span (ns >= 32, so all 32 rows belong to the same group)
        float m0 = -1e30f, m1 = -1e30f;
#pragma unroll
        for (int j = 0; j < 8; ++j) {
            m0 = fmaxf(m0, fmaxf(acc00[j] * sc0 + sh0, 0.0f));
            m0 = fmaxf(m0, fmaxf(acc10[j] * sc0 + sh0, 0.0f));
            m1 = fmaxf(m1, fmaxf(acc01[j] * sc1 + sh1, 0.0f));
            m1 = fmaxf(m1, fmaxf(acc11[j] * sc1 + sh1, 0.0f));
        }
        // combine the two EXEC halves (same columns, complementary rows)
        m0 = fmaxf(m0, __shfl_xor(m0, 16, 32));
        m1 = fmaxf(m1, __shfl_xor(m1, 16, 32));
        if (half == 0) {
            s_part[mw][nw * 32 + l16]      = m0;
            s_part[mw][nw * 32 + 16 + l16] = m1;
        }
        __syncthreads();
        // combine waves sharing a group: wpg = ns/32 waves per group
        const int wpg = ns >> 5;
        const int gpb = 4 / wpg;                 // groups per block
        const int entries = gpb * 64;
        if (tid < entries) {
            const int gl = tid >> 6, col = tid & 63;
            float v = -1e30f;
            for (int i = 0; i < wpg; ++i) v = fmaxf(v, s_part[gl * wpg + i][col]);
            const size_t g = (size_t)blockIdx.x * gpb + gl;
            out[g * N + blockIdx.y * 64 + col] = f32_to_bf16(v);
        }
    }
}

// ---------------------------------------------------------------------------
// Final 256->10 FC + log_softmax, f32 out.
__global__ void head_final_kernel(const unsigned short* __restrict__ h,  // 32 x 256
                                  const float* __restrict__ W,           // 10 x 256
                                  const float* __restrict__ bias,
                                  float* __restrict__ out) {
    const int b = threadIdx.x;    // block = 32
    if (b >= kB) return;
    float logit[10];
    float mx = -1e30f;
    for (int c = 0; c < 10; ++c) {
        float acc = bias[c];
        const float* w = W + c * 256;
        const unsigned short* hv = h + b * 256;
        for (int k = 0; k < 256; ++k) acc += bf16_to_f32(hv[k]) * w[k];
        logit[c] = acc;
        mx = fmaxf(mx, acc);
    }
    float lse = 0.0f;
    for (int c = 0; c < 10; ++c) lse += __expf(logit[c] - mx);
    lse = __logf(lse);
    for (int c = 0; c < 10; ++c) out[b * 10 + c] = logit[c] - mx - lse;
}

// ---------------------------------------------------------------------------
extern "C" void kernel_launch(void* const* d_in, const int* in_sizes, int n_in,
                              void* d_out, int out_size, void* d_ws, size_t ws_size,
                              hipStream_t stream) {
    (void)in_sizes; (void)n_in; (void)out_size; (void)ws_size;

    // --- input layout (flattened setup_inputs order) ---
    const float* xyz_in = (const float*)d_in[0];
    static const int LBASE[11] = {1, 7, 13, 19, 25, 31, 37, 43, 49, 55, 61};
    static const int LCOUT[11] = {64, 64, 128, 128, 128, 256, 256, 512, 1024, 512, 256};
    static const int LCIN[11]  = {6, 64, 64, 131, 128, 128, 259, 256, 512, 1024, 512};
    static const int LKPAD[11] = {32, 64, 64, 160, 128, 128, 288, 256, 512, 1024, 512};
    const float* fc3W = (const float*)d_in[67];
    const float* fc3b = (const float*)d_in[68];

    // --- workspace allocator ---
    char* cur = (char*)d_ws;
    auto alloc = [&](size_t bytes) -> void* {
        void* p = (void*)cur;
        cur += (bytes + 255) & ~(size_t)255;
        return p;
    };
    const size_t M1 = (size_t)kB * 512 * 32;   // 524288
    const size_t M2 = (size_t)kB * 128 * 64;   // 262144
    const size_t M3 = (size_t)kB * 128;        // 4096

    float* xyz0 = (float*)alloc((size_t)kB * kN * 3 * 4);
    int*   idx1 = (int*)alloc((size_t)kB * 512 * 4);
    float* nx1  = (float*)alloc((size_t)kB * 512 * 3 * 4);
    int*   idx2 = (int*)alloc((size_t)kB * 128 * 4);
    float* nx2  = (float*)alloc((size_t)kB * 128 * 3 * 4);
    unsigned short* G    = (unsigned short*)alloc(M2 * 160 * 2);        // max grouped buf
    unsigned short* bufX = (unsigned short*)alloc(M1 * 64 * 2);
    unsigned short* bufY = (unsigned short*)alloc(M1 * 64 * 2);
    unsigned short* P1   = (unsigned short*)alloc((size_t)kB * 512 * 128 * 2);
    unsigned short* P2   = (unsigned short*)alloc((size_t)kB * 128 * 256 * 2);
    unsigned short* H    = (unsigned short*)alloc((size_t)kB * 1024 * 2);

    unsigned short* WbP[11]; float* scP[11]; float* shP[11];
    for (int i = 0; i < 11; ++i) {
        WbP[i] = (unsigned short*)alloc((size_t)LCOUT[i] * LKPAD[i] * 2);
        scP[i] = (float*)alloc((size_t)LCOUT[i] * 4);
        shP[i] = (float*)alloc((size_t)LCOUT[i] * 4);
    }

    // --- weight / BN prep ---
    for (int i = 0; i < 11; ++i) {
        const int base = LBASE[i];
        prep_layer_kernel<<<LCOUT[i], 128, 0, stream>>>(
            (const float*)d_in[base + 0], (const float*)d_in[base + 1],
            (const float*)d_in[base + 2], (const float*)d_in[base + 3],
            (const float*)d_in[base + 4], (const float*)d_in[base + 5],
            WbP[i], scP[i], shP[i], LCIN[i], LKPAD[i]);
    }

    auto gemm = [&](const unsigned short* Ain, int layer, unsigned short* Out,
                    size_t M, int K, int Nn) {
        dim3 grid((unsigned)((M + 127) / 128), (unsigned)(Nn / 64));
        wmma_gemm_kernel<false><<<grid, 256, 0, stream>>>(
            Ain, WbP[layer], scP[layer], shP[layer], Out, (int)M, K, Nn, 0);
    };
    auto gemm_pool = [&](const unsigned short* Ain, int layer, unsigned short* Out,
                         size_t M, int K, int Nn, int ns) {
        dim3 grid((unsigned)(M / 128), (unsigned)(Nn / 64));   // M % 128 == 0
        wmma_gemm_kernel<true><<<grid, 256, 0, stream>>>(
            Ain, WbP[layer], scP[layer], shP[layer], Out, (int)M, K, Nn, ns);
    };

    // --- stage 0: transpose xyz ---
    {
        int total = kB * kN * 3;
        transpose_xyz_kernel<<<(total + 255) / 256, 256, 0, stream>>>(xyz_in, xyz0, total);
    }

    // --- SA1 ---
    fps_kernel<<<kB, 256, 0, stream>>>(xyz0, idx1, kN, 512);
    gather_xyz_kernel<<<(kB * 512 + 255) / 256, 256, 0, stream>>>(xyz0, idx1, nx1, kN, kB * 512);
    ball_group_kernel<<<kB * 512, 32, 0, stream>>>(xyz0, nx1, xyz0, nullptr, 3, G,
                                                   kN, 512, 32, 32, 0.04f);
    gemm(G,    0, bufX, M1, 32, 64);
    gemm(bufX, 1, bufY, M1, 64, 64);
    gemm_pool(bufY, 2, P1, M1, 64, 128, 32);     // -> (kB*512) x 128 pooled

    // --- SA2 ---
    fps_kernel<<<kB, 256, 0, stream>>>(nx1, idx2, 512, 128);
    gather_xyz_kernel<<<(kB * 128 + 255) / 256, 256, 0, stream>>>(nx1, idx2, nx2, 512, kB * 128);
    ball_group_kernel<<<kB * 128, 64, 0, stream>>>(nx1, nx2, nullptr, P1, 128, G,
                                                   512, 128, 64, 160, 0.16f);
    gemm(G,    3, bufX, M2, 160, 128);
    gemm(bufX, 4, bufY, M2, 128, 128);
    gemm_pool(bufY, 5, P2, M2, 128, 256, 64);    // -> (kB*128) x 256 pooled

    // --- SA3 (group_all) ---
    group_all_kernel<<<kB, 128, 0, stream>>>(nx2, P2, G, 128, 256, 288);
    gemm(G,    6, bufX, M3, 288, 256);
    gemm(bufX, 7, bufY, M3, 256, 512);
    gemm_pool(bufY, 8, H, M3, 512, 1024, 128);   // -> kB x 1024 pooled

    // --- head ---
    gemm(H,    9,  bufX, 32, 1024, 512);
    gemm(bufX, 10, bufY, 32, 512, 256);
    head_final_kernel<<<1, 32, 0, stream>>>(bufY, fc3W, fc3b, (float*)d_out);
}
